// Transformer_18227841204349
// MI455X (gfx1250) — compile-verified, matrix operand order checked by reading, not run on hardware
//
#include <hip/hip_runtime.h>
#include <hip/hip_bf16.h>

// ---------------------------------------------------------------------------
// Problem constants (from reference)
// ---------------------------------------------------------------------------
constexpr int kV = 16384;   // vocab
constexpr int kD = 512;     // embed dim
constexpr int kH = 8;       // heads
constexpr int kM = 2048;    // mlp dim
constexpr int kL = 4;       // mlp mid layers
constexpr int kB = 4;       // batch
constexpr int kS = 1024;    // seq
constexpr int kBS = kB * kS;        // 4096 rows
constexpr float kEPS = 1e-5f;

typedef __attribute__((ext_vector_type(16))) __bf16 v16bf;
typedef __attribute__((ext_vector_type(8)))  float  v8f;
typedef unsigned short u16;

// ---------------------------------------------------------------------------
// CDNA5 async global->LDS path (ASYNCcnt), guarded so either toolchain builds
// ---------------------------------------------------------------------------
#if defined(__has_builtin)
#  if __has_builtin(__builtin_amdgcn_global_load_async_to_lds_b128)
#    define ASYNC_LDS 1
#  endif
#  if __has_builtin(__builtin_amdgcn_s_wait_asynccnt)
#    define ASYNC_WAIT_BUILTIN 1
#  endif
#endif
#ifndef ASYNC_LDS
#  define ASYNC_LDS 0
#endif
#ifndef ASYNC_WAIT_BUILTIN
#  define ASYNC_WAIT_BUILTIN 0
#endif

typedef int v4i __attribute__((vector_size(16)));            // matches builtin proto
typedef __attribute__((address_space(1))) v4i* gv4i_p;       // global (__device__) int4*
typedef __attribute__((address_space(3))) v4i* lv4i_p;       // LDS (__shared__) int4*

__device__ __forceinline__ void stage16(void* lds, const void* g) {
#if ASYNC_LDS
  __builtin_amdgcn_global_load_async_to_lds_b128((gv4i_p)g, (lv4i_p)lds, 0, 0);
#else
  *(uint4*)lds = *(const uint4*)g;        // global_load_b128 + ds_store_b128
#endif
}

template <int N>
__device__ __forceinline__ void wait_async() {
#if ASYNC_LDS
#  if ASYNC_WAIT_BUILTIN
  __builtin_amdgcn_s_wait_asynccnt(N);
#  else
  asm volatile("s_wait_asynccnt %0" :: "i"(N));
#  endif
#endif
}

__device__ __forceinline__ u16 f2bf(float f) {
  union { float f; unsigned u; } x; x.f = f;
  unsigned r = x.u + 0x7FFFu + ((x.u >> 16) & 1u);   // round-to-nearest-even
  return (u16)(r >> 16);
}

// ---------------------------------------------------------------------------
// f32 -> bf16 conversion (weights / activations)
// ---------------------------------------------------------------------------
__global__ __launch_bounds__(256)
void cvt_f32_bf16(const float* __restrict__ in, u16* __restrict__ out, long long n) {
  long long i = (long long)blockIdx.x * 256 + threadIdx.x;
  long long stride = (long long)gridDim.x * 256;
  for (; i < n; i += stride) out[i] = f2bf(in[i]);
}

// ---------------------------------------------------------------------------
// Embedding + positional encoding + LayerNorm  (one block per token row)
// ---------------------------------------------------------------------------
__global__ __launch_bounds__(256)
void embed_ln(const int* __restrict__ x, const float* __restrict__ emb,
              const float* __restrict__ g, const float* __restrict__ bta,
              float* __restrict__ eF, u16* __restrict__ eB) {
  const int row = blockIdx.x;          // b*S + s
  const int s   = row % kS;
  const int tid = threadIdx.x;
  __shared__ float red[256];

  const int tok = x[row];
  constexpr float c = (float)(9.210340371976184 / (double)kD); // ln(1e4)/D
  float vals[2];
#pragma unroll
  for (int t = 0; t < 2; ++t) {
    int d  = tid + t * 256;
    int i2 = (d >> 1) << 1;
    float div = __expf(-(float)i2 * c);
    float arg = (float)s * div;
    float pe  = (d & 1) ? __cosf(arg) : __sinf(arg);
    vals[t] = emb[(long long)tok * kD + d] + pe;
  }
  red[tid] = vals[0] + vals[1];
  __syncthreads();
  for (int o = 128; o > 0; o >>= 1) { if (tid < o) red[tid] += red[tid + o]; __syncthreads(); }
  const float mu = red[0] / (float)kD;
  __syncthreads();
  float d0 = vals[0] - mu, d1 = vals[1] - mu;
  red[tid] = d0 * d0 + d1 * d1;
  __syncthreads();
  for (int o = 128; o > 0; o >>= 1) { if (tid < o) red[tid] += red[tid + o]; __syncthreads(); }
  const float rstd = rsqrtf(red[0] / (float)kD + kEPS);

#pragma unroll
  for (int t = 0; t < 2; ++t) {
    int d = tid + t * 256;
    float o = (vals[t] - mu) * rstd * g[d] + bta[d];
    long long idx = (long long)row * kD + d;
    eF[idx] = o;
    eB[idx] = f2bf(o);
  }
}

// ---------------------------------------------------------------------------
// LN(attn + e) then ReLU -> bf16 (feeds the W0 GEMM)
// ---------------------------------------------------------------------------
__global__ __launch_bounds__(256)
void ln2_relu(const float* __restrict__ attn, const float* __restrict__ eF,
              const float* __restrict__ g, const float* __restrict__ bta,
              u16* __restrict__ outB) {
  const int row = blockIdx.x;
  const int tid = threadIdx.x;
  __shared__ float red[256];

  float vals[2];
#pragma unroll
  for (int t = 0; t < 2; ++t) {
    long long idx = (long long)row * kD + tid + t * 256;
    vals[t] = attn[idx] + eF[idx];
  }
  red[tid] = vals[0] + vals[1];
  __syncthreads();
  for (int o = 128; o > 0; o >>= 1) { if (tid < o) red[tid] += red[tid + o]; __syncthreads(); }
  const float mu = red[0] / (float)kD;
  __syncthreads();
  float d0 = vals[0] - mu, d1 = vals[1] - mu;
  red[tid] = d0 * d0 + d1 * d1;
  __syncthreads();
  for (int o = 128; o > 0; o >>= 1) { if (tid < o) red[tid] += red[tid + o]; __syncthreads(); }
  const float rstd = rsqrtf(red[0] / (float)kD + kEPS);

#pragma unroll
  for (int t = 0; t < 2; ++t) {
    int d = tid + t * 256;
    float o = (vals[t] - mu) * rstd * g[d] + bta[d];
    o = fmaxf(o, 0.f);
    outB[(long long)row * kD + d] = f2bf(o);
  }
}

// ---------------------------------------------------------------------------
// Masked softmax over score rows.  mask: j < q -> -inf (tril(-1) semantics).
// ---------------------------------------------------------------------------
__global__ __launch_bounds__(256)
void softmax_mask(const float* __restrict__ scores, u16* __restrict__ w, float invs) {
  const long long row = blockIdx.x;        // h*B*S + b*S + q
  const int q   = (int)(row % kS);
  const int tid = threadIdx.x;
  const float* src = scores + row * kS;
  u16*        dst  = w      + row * kS;
  __shared__ float red[256];

  float v[4];
  float mx = -3.0e38f;
#pragma unroll
  for (int t = 0; t < 4; ++t) {
    int j = tid + t * 256;
    float s = (j < q) ? -3.0e38f : src[j] * invs;
    v[t] = s;
    mx = fmaxf(mx, s);
  }
  red[tid] = mx;
  __syncthreads();
  for (int o = 128; o > 0; o >>= 1) { if (tid < o) red[tid] = fmaxf(red[tid], red[tid + o]); __syncthreads(); }
  const float m = red[0];
  __syncthreads();

  float e[4];
  float se = 0.f;
#pragma unroll
  for (int t = 0; t < 4; ++t) {
    e[t] = (v[t] <= -2.9e38f) ? 0.f : __expf(v[t] - m);
    se += e[t];
  }
  red[tid] = se;
  __syncthreads();
  for (int o = 128; o > 0; o >>= 1) { if (tid < o) red[tid] += red[tid + o]; __syncthreads(); }
  const float inv = 1.f / red[0];
#pragma unroll
  for (int t = 0; t < 4; ++t) dst[tid + t * 256] = f2bf(e[t] * inv);
}

// ---------------------------------------------------------------------------
// Batched bf16 WMMA GEMM:  C[M,N] = scale * (A[M,K] @ Bw[N,K]^T) (+bias)(ReLU)(+Res)
//   256 threads = 8 wave32, block tile 128x128, K-step 32, double-buffered LDS
//   with async global->LDS staging pipelined against v_wmma_f32_16x16x32_bf16.
//   batch z = z1*z2ext + z2; Rred = in-kernel reduction batches (head sum).
//   Requires: K a multiple of 32 with K/32 a power of two (true for all uses).
// Epilogue flags: 1=bias  2=relu  4=residual(f32)  8=non-temporal f32 store
// ---------------------------------------------------------------------------
__global__ __launch_bounds__(256)
void gemm_bf16_wmma(const u16* __restrict__ A, const u16* __restrict__ Bw,
                    const float* __restrict__ bias, const float* __restrict__ Res,
                    float* __restrict__ OutF, u16* __restrict__ OutB, u16* __restrict__ OutT,
                    int Kdim,
                    int lda, int ldb, int ldf, int ldob, int ldot, int ldres,
                    int z2ext,
                    long long sA1, long long sA2,
                    long long sB1, long long sB2,
                    long long sF1, long long sF2,
                    long long sOB1, long long sOB2,
                    long long sOT1, long long sOT2,
                    long long sBias1,
                    int Rred, long long rsA, long long rsB,
                    int flags, float scale) {
  const int z  = blockIdx.z;
  const int z1 = z / z2ext, z2 = z % z2ext;
  A  += z1 * sA1 + z2 * sA2;
  Bw += z1 * sB1 + z2 * sB2;
  if (bias) bias += z1 * sBias1;
  if (OutF) OutF += z1 * sF1 + z2 * sF2;
  if (OutB) OutB += z1 * sOB1 + z2 * sOB2;
  if (OutT) OutT += z1 * sOT1 + z2 * sOT2;

  constexpr int LDK = 40;                 // 32 + 16B pad -> conflict-free b128 ds reads
  __shared__ alignas(16) u16 As[2][128 * LDK];
  __shared__ alignas(16) u16 Bs[2][128 * LDK];

  const int tid  = threadIdx.x;
  const int lane = tid & 31;
  const int wave = tid >> 5;
  const int wm   = wave >> 2;             // 0..1
  const int wn   = wave & 3;              // 0..3
  const int half = lane >> 4;             // 0|1 (K sub-stripe)
  const int lr   = lane & 15;
  const int m0   = blockIdx.y * 128;
  const int n0   = blockIdx.x * 128;

  const int ldRow = tid >> 2;             // 0..63
  const int ldCol = (tid & 3) * 8;        // bf16 column chunk (16B)

  v8f acc[4][2] = {};
  union Frag { uint4 u[2]; v16bf v; };

  const int nk      = Kdim >> 5;          // K tiles (power of two for all uses)
  const int nkShift = __builtin_ctz(nk);
  const int nkMask  = nk - 1;
  const int T       = Rred << nkShift;    // total pipeline tiles

  auto stage_tile = [&](int buf, int t) {
    const int r  = t >> nkShift;
    const int k0 = (t & nkMask) << 5;
    const u16* Ar = A  + (long long)r * rsA;
    const u16* Br = Bw + (long long)r * rsB;
    stage16(&As[buf][ldRow * LDK + ldCol],
            Ar + (long long)(m0 + ldRow) * lda + k0 + ldCol);
    stage16(&As[buf][(ldRow + 64) * LDK + ldCol],
            Ar + (long long)(m0 + ldRow + 64) * lda + k0 + ldCol);
    stage16(&Bs[buf][ldRow * LDK + ldCol],
            Br + (long long)(n0 + ldRow) * ldb + k0 + ldCol);
    stage16(&Bs[buf][(ldRow + 64) * LDK + ldCol],
            Br + (long long)(n0 + ldRow + 64) * ldb + k0 + ldCol);
  };

  auto compute_tile = [&](int cur) {
    Frag a[4], b[2];
#pragma unroll
    for (int i = 0; i < 4; ++i) {
      const u16* p = &As[cur][(wm * 64 + i * 16 + lr) * LDK + half * 8];
      a[i].u[0] = *(const uint4*)p;        // K {0..7}  | {8..15}
      a[i].u[1] = *(const uint4*)(p + 16); // K {16..23}| {24..31}
    }
#pragma unroll
    for (int j = 0; j < 2; ++j) {
      const u16* p = &Bs[cur][(wn * 32 + j * 16 + lr) * LDK + half * 8];
      b[j].u[0] = *(const uint4*)p;
      b[j].u[1] = *(const uint4*)(p + 16);
    }
#pragma unroll
    for (int i = 0; i < 4; ++i)
#pragma unroll
      for (int j = 0; j < 2; ++j)
        acc[i][j] = __builtin_amdgcn_wmma_f32_16x16x32_bf16(
            false, a[i].v, false, b[j].v, (short)0, acc[i][j], false, false);
  };

  stage_tile(0, 0);
  int t = 0;
#pragma unroll 2
  for (; t < T - 1; ++t) {                // steady state: always prefetch + wait 4
    const int cur = t & 1;
    stage_tile(cur ^ 1, t + 1);
    wait_async<4>();                      // previous tile's 4 loads retired
    __syncthreads();                      // all waves' current tile visible
    compute_tile(cur);
    __syncthreads();                      // reads done before buffer reuse
  }
  wait_async<0>();                        // peeled last tile: drain everything
  __syncthreads();
  compute_tile(t & 1);

  // epilogue: lane holds (m = rr + 8*half, n = lr) of each 16x16 tile
#pragma unroll
  for (int i = 0; i < 4; ++i) {
#pragma unroll
    for (int j = 0; j < 2; ++j) {
      const int n = n0 + wn * 32 + j * 16 + lr;
      const float bv = (flags & 1) ? bias[n] : 0.f;
#pragma unroll
      for (int rr = 0; rr < 8; ++rr) {
        const int m = m0 + wm * 64 + i * 16 + half * 8 + rr;
        float v = acc[i][j][rr] * scale + bv;
        if (flags & 2) v = fmaxf(v, 0.f);
        if (flags & 4) v += Res[(long long)m * ldres + n];
        if (OutF) {
          float* p = &OutF[(long long)m * ldf + n];
          if (flags & 8) __builtin_nontemporal_store(v, p);   // logits: skip L2 retention
          else *p = v;
        }
        if (OutB) OutB[(long long)m * ldob + n] = f2bf(v);
        if (OutT) OutT[(long long)n * ldot + m] = f2bf(v);    // transposed store (V^T)
      }
    }
  }
}

// ---------------------------------------------------------------------------
// host side
// ---------------------------------------------------------------------------
extern "C" void kernel_launch(void* const* d_in, const int* in_sizes, int n_in,
                              void* d_out, int out_size, void* d_ws, size_t ws_size,
                              hipStream_t stream) {
  (void)in_sizes; (void)n_in; (void)out_size; (void)ws_size;

  const int*   x    = (const int*)  d_in[0];
  const float* emb  = (const float*)d_in[1];
  const float* Wq   = (const float*)d_in[2];
  const float* bq   = (const float*)d_in[3];
  const float* Wk   = (const float*)d_in[4];
  const float* bk   = (const float*)d_in[5];
  const float* Wv   = (const float*)d_in[6];
  const float* bv   = (const float*)d_in[7];
  const float* ln_g = (const float*)d_in[8];
  const float* ln_b = (const float*)d_in[9];
  const float* W0   = (const float*)d_in[10];
  const float* b0   = (const float*)d_in[11];
  const float* Wmid = (const float*)d_in[12];
  const float* bmid = (const float*)d_in[13];
  const float* Wout = (const float*)d_in[14];
  const float* bout = (const float*)d_in[15];
  float* out = (float*)d_out;

  char* ws = (char*)d_ws;
  size_t off = 0;
  auto alloc = [&](size_t bytes) -> void* {
    void* p = ws + off;
    off = (off + bytes + 255) & ~(size_t)255;
    return p;
  };

  const long long nQKVW = (long long)kH * kD * kD;
  const long long nE    = (long long)kBS * kD;
  const long long nQKV  = (long long)kH * kB * kS * kD;
  const long long nSc   = (long long)kH * kB * kS * kS;
  const long long nH    = (long long)kBS * kM;

  float* eF     = (float*)alloc(nE * 4);
  u16*   eB     = (u16*)  alloc(nE * 2);
  u16*   WqB    = (u16*)  alloc(nQKVW * 2);
  u16*   WkB    = (u16*)  alloc(nQKVW * 2);
  u16*   WvB    = (u16*)  alloc(nQKVW * 2);
  u16*   W0B    = (u16*)  alloc((long long)kM * kD * 2);
  u16*   WmidB  = (u16*)  alloc((long long)kL * kM * kM * 2);
  u16*   WoutB  = (u16*)  alloc((long long)kV * kM * 2);
  u16*   qB     = (u16*)  alloc(nQKV * 2);
  u16*   kB16   = (u16*)  alloc(nQKV * 2);
  u16*   vTB    = (u16*)  alloc(nQKV * 2);   // [h,b,d,s]
  float* scF    = (float*)alloc(nSc * 4);
  u16*   wB     = (u16*)  alloc(nSc * 2);
  float* attnF  = (float*)alloc(nE * 4);
  u16*   ln2B   = (u16*)  alloc(nE * 2);
  float* hFa    = (float*)alloc(nH * 4);
  u16*   hBa    = (u16*)  alloc(nH * 2);
  float* hFb    = (float*)alloc(nH * 4);
  u16*   hBb    = (u16*)  alloc(nH * 2);

  auto cvt = [&](const float* src, u16* dst, long long n) {
    int blocks = (int)((n + 255) / 256);
    if (blocks > 2048) blocks = 2048;
    cvt_f32_bf16<<<blocks, 256, 0, stream>>>(src, dst, n);
  };
  cvt(Wq,   WqB,   nQKVW);
  cvt(Wk,   WkB,   nQKVW);
  cvt(Wv,   WvB,   nQKVW);
  cvt(W0,   W0B,   (long long)kM * kD);
  cvt(Wmid, WmidB, (long long)kL * kM * kM);
  cvt(Wout, WoutB, (long long)kV * kM);

  embed_ln<<<kBS, 256, 0, stream>>>(x, emb, ln_g, ln_b, eF, eB);

  const long long sBSD = (long long)kS * kD;   // per-b activation stride
  const long long sHD2 = (long long)kD * kD;   // per-head weight stride

  // ---- Q projection ----
  gemm_bf16_wmma<<<dim3(kD / 128, kS / 128, kH * kB), 256, 0, stream>>>(
      eB, WqB, bq, nullptr, nullptr, qB, nullptr,
      kD, kD, kD, 0, kD, 0, 0, kB,
      0, sBSD, sHD2, 0, 0, 0,
      (long long)kB * sBSD, sBSD, 0, 0, kD,
      1, 0, 0, /*flags=*/1, 1.0f);
  // ---- K projection ----
  gemm_bf16_wmma<<<dim3(kD / 128, kS / 128, kH * kB), 256, 0, stream>>>(
      eB, WkB, bk, nullptr, nullptr, kB16, nullptr,
      kD, kD, kD, 0, kD, 0, 0, kB,
      0, sBSD, sHD2, 0, 0, 0,
      (long long)kB * sBSD, sBSD, 0, 0, kD,
      1, 0, 0, /*flags=*/1, 1.0f);
  // ---- V projection, stored transposed vT[h,b,d,s] ----
  gemm_bf16_wmma<<<dim3(kD / 128, kS / 128, kH * kB), 256, 0, stream>>>(
      eB, WvB, bv, nullptr, nullptr, nullptr, vTB,
      kD, kD, kD, 0, 0, kS, 0, kB,
      0, sBSD, sHD2, 0, 0, 0,
      0, 0, (long long)kB * kD * kS, (long long)kD * kS, kD,
      1, 0, 0, /*flags=*/1, 1.0f);

  // ---- scores = q @ k^T (f32) ----
  gemm_bf16_wmma<<<dim3(kS / 128, kS / 128, kH * kB), 256, 0, stream>>>(
      qB, kB16, nullptr, nullptr, scF, nullptr, nullptr,
      kD, kD, kD, kS, 0, 0, 0, kB,
      (long long)kB * sBSD, sBSD, (long long)kB * sBSD, sBSD,
      (long long)kB * kS * kS, (long long)kS * kS, 0, 0, 0, 0, 0,
      1, 0, 0, /*flags=*/0, 1.0f);

  // ---- masked softmax (1/sqrt(D) folded in), bf16 weights ----
  softmax_mask<<<kH * kB * kS, 256, 0, stream>>>(scF, wB, 1.0f / sqrtf((float)kD));

  // ---- attn = sum_h (w[h,b] @ v[h,b]) -> f32 [b,s,d]; head sum via Rred ----
  gemm_bf16_wmma<<<dim3(kD / 128, kS / 128, kB), 256, 0, stream>>>(
      wB, vTB, nullptr, nullptr, attnF, nullptr, nullptr,
      kS, kS, kS, kD, 0, 0, 0, kB,
      0, (long long)kS * kS, 0, (long long)kD * kS,
      0, sBSD, 0, 0, 0, 0, 0,
      /*Rred=*/kH, (long long)kB * kS * kS, (long long)kB * kD * kS,
      /*flags=*/0, 1.0f);

  // ---- h = relu(LN(attn + e)) ----
  ln2_relu<<<kBS, 256, 0, stream>>>(attnF, eF, ln_g, ln_b, ln2B);

  // ---- h0 = h @ W0^T + b0 ----
  gemm_bf16_wmma<<<dim3(kM / 128, kBS / 128, 1), 256, 0, stream>>>(
      ln2B, W0B, b0, nullptr, hFa, hBa, nullptr,
      kD, kD, kD, kM, kM, 0, 0, 1,
      0, 0, 0, 0, 0, 0, 0, 0, 0, 0, 0,
      1, 0, 0, /*flags=*/1, 1.0f);

  // ---- mid layers: h = relu(h @ Wmid[i]^T + bmid[i]) + h ----
  float* hFin = hFa; u16* hBin = hBa;
  float* hFot = hFb; u16* hBot = hBb;
  for (int i = 0; i < kL; ++i) {
    gemm_bf16_wmma<<<dim3(kM / 128, kBS / 128, 1), 256, 0, stream>>>(
        hBin, WmidB + (long long)i * kM * kM, bmid + (long long)i * kM, hFin,
        hFot, hBot, nullptr,
        kM, kM, kM, kM, kM, 0, kM, 1,
        0, 0, 0, 0, 0, 0, 0, 0, 0, 0, 0,
        1, 0, 0, /*flags=*/7, 1.0f);
    float* tf = hFin; hFin = hFot; hFot = tf;
    u16*   tb = hBin; hBin = hBot; hBot = tb;
  }

  // ---- logits = h @ Wout^T + bout -> d_out [B,S,V] f32 (non-temporal) ----
  gemm_bf16_wmma<<<dim3(kV / 128, kBS / 128, 1), 256, 0, stream>>>(
      hBin, WoutB, bout, nullptr, out, nullptr, nullptr,
      kM, kM, kM, kV, 0, 0, 0, 1,
      0, 0, 0, 0, 0, 0, 0, 0, 0, 0, 0,
      1, 0, 0, /*flags=*/9, 1.0f);
}
